// LSTMCell_38414187495767
// MI455X (gfx1250) — matrix operand order repaired and checked by reading, not compiled, not run on hardware
//
#include <hip/hip_runtime.h>
#include <hip/hip_bf16.h>
#include <math.h>

// ---------------------------------------------------------------------------
// LSTM cell for gfx1250 (MI455X), wave32 + WMMA bf16 "3xBF16" split path.
//   gates = [x | h0] @ [Wi ; Wh] + b   (M=4096, N=4*1024, K=2048)
//
// Fast path (ws_size >= 64MB):
//   1) split_pack:    x,h0 -> A_hi/A_lo bf16 [4096][2048]   (RNE hi + residual lo)
//   2) transpose_split: W  -> Bt_hi/Bt_lo bf16 [gate][n][2048] (k-contiguous)
//   3) lstm_gemm_packed: staging is pure b128 global->LDS copies (async-to-LDS
//      when the builtin exists), 24 v_wmma_f32_16x16x32_bf16 per K-step,
//      fused bias+sigmoid/tanh+cell-update epilogue.
// Fallback path: round-1 fused kernel (inline f32->bf16 split during staging).
// ---------------------------------------------------------------------------

typedef __attribute__((ext_vector_type(16))) __bf16 v16bf;
typedef __attribute__((ext_vector_type(8)))  __bf16 bf8;   // 16 bytes
typedef __attribute__((ext_vector_type(4)))  __bf16 bf4;   // 8 bytes
typedef __attribute__((ext_vector_type(8)))  float  v8f;
typedef int v4i_vs __attribute__((__vector_size__(16)));   // matches builtin param

#define HID       1024
#define BATCH     4096
#define BM        128      // M tile
#define NH        32       // H-columns per block
#define BN        (4*NH)   // 128 gate columns per block
#define KTILE     32
#define KSTEPS    (2048/KTILE)
#define LDSTRIDE  40       // bf16 elems; 80B row = 16B aligned, low conflicts

#ifndef __has_builtin
#define __has_builtin(x) 0
#endif
#if __has_builtin(__builtin_amdgcn_global_load_async_to_lds_b128) && \
    __has_builtin(__builtin_amdgcn_s_wait_asynccnt)
#define HAVE_ASYNC_LDS 1
#else
#define HAVE_ASYNC_LDS 0
#endif

#define AS1 __attribute__((address_space(1)))
#define AS3 __attribute__((address_space(3)))

__device__ __forceinline__ float sigmoid_f(float z) {
    return 1.0f / (1.0f + __expf(-z));
}
__device__ __forceinline__ float tanh_f(float z) {
    z = fminf(fmaxf(z, -15.0f), 15.0f);
    float e = __expf(2.0f * z);
    return (e - 1.0f) / (e + 1.0f);
}

// 16-byte global -> LDS copy; async-to-LDS path on CDNA5 when available.
__device__ __forceinline__ void copy16_g2l(const __bf16* __restrict__ g, __bf16* l) {
#if HAVE_ASYNC_LDS
    // two-step casts: addrspace cast (same pointee), then pointee bitcast
    AS1 __bf16* g1 = (AS1 __bf16*)g;      // C-style cast drops const + AS cast
    AS3 __bf16* l3 = (AS3 __bf16*)l;
    __builtin_amdgcn_global_load_async_to_lds_b128(
        (AS1 v4i_vs*)g1, (AS3 v4i_vs*)l3, 0, 0);
#else
    *(bf8*)l = *(const bf8*)g;
#endif
}

// ---------------------------------------------------------------------------
// Prepack kernel 1: elementwise f32 -> bf16 hi/lo split.
// src is [4096][1024]; dst rows have stride 2048 with column offset colOff
// (x -> cols [0,1024), h0 -> cols [1024,2048)).
// ---------------------------------------------------------------------------
__global__ __launch_bounds__(256)
void split_pack_kernel(const float* __restrict__ src,
                       __bf16* __restrict__ hi, __bf16* __restrict__ lo,
                       int colOff)
{
    const size_t idx = (size_t)blockIdx.x * 256 + threadIdx.x;
    const size_t f   = idx * 4;                 // flat float index
    const int    m   = (int)(f >> 10);
    const int    c   = (int)(f & 1023);
    const float4 v   = *(const float4*)(src + f);
    const float vv[4] = {v.x, v.y, v.z, v.w};
    bf4 h4, l4;
    #pragma unroll
    for (int j = 0; j < 4; ++j) {
        __bf16 hb = (__bf16)vv[j];
        h4[j] = hb;
        l4[j] = (__bf16)(vv[j] - (float)hb);
    }
    const size_t d = (size_t)m * 2048 + colOff + c;
    *(bf4*)(hi + d) = h4;
    *(bf4*)(lo + d) = l4;
}

// ---------------------------------------------------------------------------
// Prepack kernel 2: W [1024 k][1024 n] f32 -> Bt [n][2048 k] bf16 hi/lo,
// transposed via a 32x32 LDS tile. colOff = 0 (Wi side) or 1024 (Wh side).
// ---------------------------------------------------------------------------
__global__ __launch_bounds__(256)
void transpose_split_kernel(const float* __restrict__ W,
                            __bf16* __restrict__ outHi, __bf16* __restrict__ outLo,
                            int colOff)
{
    __shared__ float tile[32][33];
    const int t     = threadIdx.x;
    const int kBase = blockIdx.x * 32;
    const int nBase = blockIdx.y * 32;

    {   // load 32x32 f32 tile (coalesced along n)
        const int r  = t >> 3;
        const int c4 = (t & 7) << 2;
        const float4 v = *(const float4*)(W + (size_t)(kBase + r) * HID + nBase + c4);
        tile[r][c4 + 0] = v.x;
        tile[r][c4 + 1] = v.y;
        tile[r][c4 + 2] = v.z;
        tile[r][c4 + 3] = v.w;
    }
    __syncthreads();
    {   // write transposed, split to bf16 hi/lo (contiguous along k)
        const int nl = t >> 3;
        const int k4 = (t & 7) << 2;
        bf4 h4, l4;
        #pragma unroll
        for (int j = 0; j < 4; ++j) {
            const float xv = tile[k4 + j][nl];
            __bf16 hb = (__bf16)xv;
            h4[j] = hb;
            l4[j] = (__bf16)(xv - (float)hb);
        }
        const size_t d = (size_t)(nBase + nl) * 2048 + colOff + kBase + k4;
        *(bf4*)(outHi + d) = h4;
        *(bf4*)(outLo + d) = l4;
    }
}

// ---------------------------------------------------------------------------
// Main GEMM on prepacked bf16: staging is pure b128 copies, no conversion VALU.
// ---------------------------------------------------------------------------
__global__ __launch_bounds__(256)
void lstm_gemm_packed(
    const __bf16* __restrict__ Ahi, const __bf16* __restrict__ Alo,
    const __bf16* __restrict__ Bthi, const __bf16* __restrict__ Btlo,
    const float* __restrict__ c0,
    const float* __restrict__ b_ii, const float* __restrict__ b_if,
    const float* __restrict__ b_ig, const float* __restrict__ b_io,
    const float* __restrict__ b_hi, const float* __restrict__ b_hf,
    const float* __restrict__ b_hg, const float* __restrict__ b_ho,
    float* __restrict__ h1, float* __restrict__ c1)
{
    __shared__ __bf16 As_hi[BM * LDSTRIDE];
    __shared__ __bf16 As_lo[BM * LDSTRIDE];
    __shared__ __bf16 Bs_hi[BN * LDSTRIDE];
    __shared__ __bf16 Bs_lo[BN * LDSTRIDE];

    const int t    = threadIdx.x;
    const int w    = t >> 5;
    const int l    = t & 31;
    const int half = l >> 4;
    const int lm   = l & 15;

    const int mBase  = (blockIdx.x >> 5) * BM;
    const int nhBase = (blockIdx.x & 31) * NH;

    v8f acc[8];
    #pragma unroll
    for (int i = 0; i < 8; ++i)
        acc[i] = (v8f){0.f, 0.f, 0.f, 0.f, 0.f, 0.f, 0.f, 0.f};

    for (int kb = 0; kb < KSTEPS; ++kb) {
        __syncthreads();   // previous iteration's LDS reads are done
        const int kG = kb * KTILE;

        // ---- staging: 8 x b128 global->LDS copies per thread --------------
        // 512 16B segments per tile; idx -> row = idx>>2, elem-off = (idx&3)*8
        #pragma unroll
        for (int i = 0; i < 2; ++i) {
            const int idx = t + (i << 8);
            const int row = idx >> 2;            // 0..127
            const int sg  = (idx & 3) << 3;      // 0,8,16,24 (bf16 elems)
            const int lo_ = row * LDSTRIDE + sg;

            // A tile: rows mBase..mBase+127, k = kG..kG+31
            const size_t ga = (size_t)(mBase + row) * 2048 + kG + sg;
            copy16_g2l(Ahi + ga, &As_hi[lo_]);
            copy16_g2l(Alo + ga, &As_lo[lo_]);

            // B tile (transposed layout): row = gate*32 + col
            const int g  = row >> 5;
            const int cc = row & 31;
            const size_t gb = ((size_t)(g << 10) + nhBase + cc) * 2048 + kG + sg;
            copy16_g2l(Bthi + gb, &Bs_hi[lo_]);
            copy16_g2l(Btlo + gb, &Bs_lo[lo_]);
        }
#if HAVE_ASYNC_LDS
        __builtin_amdgcn_s_wait_asynccnt(0);
#endif
        __syncthreads();

        // ---- fragments + WMMA (wave w owns M rows [16w, 16w+16)) ----------
        // A 16-bit 16x32 layout: lane half h, element e: k = 8h+e | 16+8h+(e-8)
        const int aOff = (16 * w + lm) * LDSTRIDE + 8 * half;
        const bf8 a0h = *(const bf8*)&As_hi[aOff];
        const bf8 a1h = *(const bf8*)&As_hi[aOff + 16];
        const bf8 a0l = *(const bf8*)&As_lo[aOff];
        const bf8 a1l = *(const bf8*)&As_lo[aOff + 16];
        const v16bf a_hi = __builtin_shufflevector(a0h, a1h,
            0,1,2,3,4,5,6,7,8,9,10,11,12,13,14,15);
        const v16bf a_lo = __builtin_shufflevector(a0l, a1l,
            0,1,2,3,4,5,6,7,8,9,10,11,12,13,14,15);

        #pragma unroll
        for (int tt = 0; tt < 8; ++tt) {
            // B 32x16 layout: lane half h, element e: k = 8h+e | 16+8h+(e-8)
            const int bOff = (16 * tt + lm) * LDSTRIDE + 8 * half;
            const bf8 b0h = *(const bf8*)&Bs_hi[bOff];
            const bf8 b1h = *(const bf8*)&Bs_hi[bOff + 16];
            const bf8 b0l = *(const bf8*)&Bs_lo[bOff];
            const bf8 b1l = *(const bf8*)&Bs_lo[bOff + 16];
            const v16bf b_h = __builtin_shufflevector(b0h, b1h,
                0,1,2,3,4,5,6,7,8,9,10,11,12,13,14,15);
            const v16bf b_l = __builtin_shufflevector(b0l, b1l,
                0,1,2,3,4,5,6,7,8,9,10,11,12,13,14,15);

            acc[tt] = __builtin_amdgcn_wmma_f32_16x16x32_bf16(
                          false, a_hi, false, b_h, (short)0, acc[tt], false, false);
            acc[tt] = __builtin_amdgcn_wmma_f32_16x16x32_bf16(
                          false, a_hi, false, b_l, (short)0, acc[tt], false, false);
            acc[tt] = __builtin_amdgcn_wmma_f32_16x16x32_bf16(
                          false, a_lo, false, b_h, (short)0, acc[tt], false, false);
        }
    }

    // ---- fused epilogue ---------------------------------------------------
    // C/D layout: acc[tt][r] -> M = 16w + 8*half + r, N = 16*tt + lm
    #pragma unroll
    for (int p = 0; p < 2; ++p) {
        const int col = nhBase + 16 * p + lm;
        const float bi = b_ii[col] + b_hi[col];
        const float bf = b_if[col] + b_hf[col];
        const float bg = b_ig[col] + b_hg[col];
        const float bo = b_io[col] + b_ho[col];
        #pragma unroll
        for (int r = 0; r < 8; ++r) {
            const int m = mBase + 16 * w + 8 * half + r;
            const float zi = acc[0 + p][r] + bi;
            const float zf = acc[2 + p][r] + bf;
            const float zg = acc[4 + p][r] + bg;
            const float zo = acc[6 + p][r] + bo;
            const float ig = sigmoid_f(zi);
            const float fg = sigmoid_f(zf);
            const float gg = tanh_f(zg);
            const float og = sigmoid_f(zo);
            const size_t idx = (size_t)m * HID + col;
            const float cv = fg * c0[idx] + ig * gg;
            c1[idx] = cv;
            h1[idx] = og * cv;
        }
    }
}

// ---------------------------------------------------------------------------
// Fallback: round-1 fused kernel (used when ws_size < 64MB).
// ---------------------------------------------------------------------------
__global__ __launch_bounds__(256)
void lstm_cell_wmma_fused(
    const float* __restrict__ x,   const float* __restrict__ h0,
    const float* __restrict__ c0,
    const float* __restrict__ Wii, const float* __restrict__ b_ii,
    const float* __restrict__ Wif, const float* __restrict__ b_if,
    const float* __restrict__ Wig, const float* __restrict__ b_ig,
    const float* __restrict__ Wio, const float* __restrict__ b_io,
    const float* __restrict__ Whi, const float* __restrict__ b_hi,
    const float* __restrict__ Whf, const float* __restrict__ b_hf,
    const float* __restrict__ Whg, const float* __restrict__ b_hg,
    const float* __restrict__ Who, const float* __restrict__ b_ho,
    float* __restrict__ h1, float* __restrict__ c1)
{
    __shared__ __bf16 As_hi[BM * LDSTRIDE];
    __shared__ __bf16 As_lo[BM * LDSTRIDE];
    __shared__ __bf16 Bs_hi[BN * LDSTRIDE];
    __shared__ __bf16 Bs_lo[BN * LDSTRIDE];

    const int t    = threadIdx.x;
    const int w    = t >> 5;
    const int l    = t & 31;
    const int half = l >> 4;
    const int lm   = l & 15;

    const int mBase  = (blockIdx.x >> 5) * BM;
    const int nhBase = (blockIdx.x & 31) * NH;

    v8f acc[8];
    #pragma unroll
    for (int i = 0; i < 8; ++i)
        acc[i] = (v8f){0.f, 0.f, 0.f, 0.f, 0.f, 0.f, 0.f, 0.f};

    for (int kb = 0; kb < KSTEPS; ++kb) {
        __syncthreads();
        const int   kOff = (kb & 31) * KTILE;
        const float* Asrc = (kb < 32) ? x : h0;
        const float* Wg[4];
        if (kb < 32) { Wg[0]=Wii; Wg[1]=Wif; Wg[2]=Wig; Wg[3]=Wio; }
        else         { Wg[0]=Whi; Wg[1]=Whf; Wg[2]=Whg; Wg[3]=Who; }

        #pragma unroll
        for (int i = 0; i < 4; ++i) {
            const int q   = t + 256 * i;
            const int row = q >> 3;
            const int c4  = (q & 7) << 2;
            const float4 v = *(const float4*)(Asrc + (size_t)(mBase + row) * HID + kOff + c4);
            const float vv[4] = {v.x, v.y, v.z, v.w};
            bf4 hi4, lo4;
            #pragma unroll
            for (int j = 0; j < 4; ++j) {
                __bf16 hb = (__bf16)vv[j];
                hi4[j] = hb;
                lo4[j] = (__bf16)(vv[j] - (float)hb);
            }
            *(bf4*)&As_hi[row * LDSTRIDE + c4] = hi4;
            *(bf4*)&As_lo[row * LDSTRIDE + c4] = lo4;
        }
        {
            const int kRow = t >> 3;
            const int c4   = (t & 7) << 2;
            #pragma unroll
            for (int g = 0; g < 4; ++g) {
                const float4 v = *(const float4*)(Wg[g] + (size_t)(kOff + kRow) * HID + nhBase + c4);
                const float vv[4] = {v.x, v.y, v.z, v.w};
                #pragma unroll
                for (int j = 0; j < 4; ++j) {
                    const int n = g * NH + c4 + j;
                    __bf16 hb = (__bf16)vv[j];
                    Bs_hi[n * LDSTRIDE + kRow] = hb;
                    Bs_lo[n * LDSTRIDE + kRow] = (__bf16)(vv[j] - (float)hb);
                }
            }
        }
        __syncthreads();

        const int aOff = (16 * w + lm) * LDSTRIDE + 8 * half;
        const bf8 a0h = *(const bf8*)&As_hi[aOff];
        const bf8 a1h = *(const bf8*)&As_hi[aOff + 16];
        const bf8 a0l = *(const bf8*)&As_lo[aOff];
        const bf8 a1l = *(const bf8*)&As_lo[aOff + 16];
        const v16bf a_hi = __builtin_shufflevector(a0h, a1h,
            0,1,2,3,4,5,6,7,8,9,10,11,12,13,14,15);
        const v16bf a_lo = __builtin_shufflevector(a0l, a1l,
            0,1,2,3,4,5,6,7,8,9,10,11,12,13,14,15);

        #pragma unroll
        for (int tt = 0; tt < 8; ++tt) {
            const int bOff = (16 * tt + lm) * LDSTRIDE + 8 * half;
            const bf8 b0h = *(const bf8*)&Bs_hi[bOff];
            const bf8 b1h = *(const bf8*)&Bs_hi[bOff + 16];
            const bf8 b0l = *(const bf8*)&Bs_lo[bOff];
            const bf8 b1l = *(const bf8*)&Bs_lo[bOff + 16];
            const v16bf b_h = __builtin_shufflevector(b0h, b1h,
                0,1,2,3,4,5,6,7,8,9,10,11,12,13,14,15);
            const v16bf b_l = __builtin_shufflevector(b0l, b1l,
                0,1,2,3,4,5,6,7,8,9,10,11,12,13,14,15);

            acc[tt] = __builtin_amdgcn_wmma_f32_16x16x32_bf16(
                          false, a_hi, false, b_h, (short)0, acc[tt], false, false);
            acc[tt] = __builtin_amdgcn_wmma_f32_16x16x32_bf16(
                          false, a_hi, false, b_l, (short)0, acc[tt], false, false);
            acc[tt] = __builtin_amdgcn_wmma_f32_16x16x32_bf16(
                          false, a_lo, false, b_h, (short)0, acc[tt], false, false);
        }
    }

    #pragma unroll
    for (int p = 0; p < 2; ++p) {
        const int col = nhBase + 16 * p + lm;
        const float bi = b_ii[col] + b_hi[col];
        const float bf = b_if[col] + b_hf[col];
        const float bg = b_ig[col] + b_hg[col];
        const float bo = b_io[col] + b_ho[col];
        #pragma unroll
        for (int r = 0; r < 8; ++r) {
            const int m = mBase + 16 * w + 8 * half + r;
            const float zi = acc[0 + p][r] + bi;
            const float zf = acc[2 + p][r] + bf;
            const float zg = acc[4 + p][r] + bg;
            const float zo = acc[6 + p][r] + bo;
            const float ig = sigmoid_f(zi);
            const float fg = sigmoid_f(zf);
            const float gg = tanh_f(zg);
            const float og = sigmoid_f(zo);
            const size_t idx = (size_t)m * HID + col;
            const float cv = fg * c0[idx] + ig * gg;
            c1[idx] = cv;
            h1[idx] = og * cv;
        }
    }
}

extern "C" void kernel_launch(void* const* d_in, const int* in_sizes, int n_in,
                              void* d_out, int out_size, void* d_ws, size_t ws_size,
                              hipStream_t stream) {
    (void)in_sizes; (void)n_in; (void)out_size;
    const float* x    = (const float*)d_in[0];
    const float* h0   = (const float*)d_in[1];
    const float* c0   = (const float*)d_in[2];
    const float* Wii  = (const float*)d_in[3];
    const float* b_ii = (const float*)d_in[4];
    const float* Wif  = (const float*)d_in[5];
    const float* b_if = (const float*)d_in[6];
    const float* Wig  = (const float*)d_in[7];
    const float* b_ig = (const float*)d_in[8];
    const float* Wio  = (const float*)d_in[9];
    const float* b_io = (const float*)d_in[10];
    const float* Whi  = (const float*)d_in[11];
    const float* b_hi = (const float*)d_in[12];
    const float* Whf  = (const float*)d_in[13];
    const float* b_hf = (const float*)d_in[14];
    const float* Whg  = (const float*)d_in[15];
    const float* b_hg = (const float*)d_in[16];
    const float* Who  = (const float*)d_in[17];
    const float* b_ho = (const float*)d_in[18];

    float* out = (float*)d_out;
    float* h1  = out;
    float* c1  = out + (size_t)BATCH * HID;

    const size_t elemsA   = (size_t)BATCH * 2048;        // 8M bf16 = 16MB
    const size_t elemsB   = (size_t)4 * HID * 2048;      // 8M bf16 = 16MB
    const size_t wsNeeded = (2 * elemsA + 2 * elemsB) * sizeof(__bf16);  // 64MB

    if (ws_size >= wsNeeded) {
        __bf16* Ahi  = (__bf16*)d_ws;
        __bf16* Alo  = Ahi + elemsA;
        __bf16* Bthi = Alo + elemsA;
        __bf16* Btlo = Bthi + elemsB;

        // 1) split x|h0 into bf16 hi/lo, layout [4096][2048]
        split_pack_kernel<<<4096, 256, 0, stream>>>(x,  Ahi, Alo, 0);
        split_pack_kernel<<<4096, 256, 0, stream>>>(h0, Ahi, Alo, 1024);

        // 2) transpose+split weights into [gate][n][2048 k]
        const float* Wi[4] = {Wii, Wif, Wig, Wio};
        const float* Wh[4] = {Whi, Whf, Whg, Who};
        for (int g = 0; g < 4; ++g) {
            __bf16* oh = Bthi + (size_t)g * HID * 2048;
            __bf16* ol = Btlo + (size_t)g * HID * 2048;
            transpose_split_kernel<<<dim3(32, 32), 256, 0, stream>>>(Wi[g], oh, ol, 0);
            transpose_split_kernel<<<dim3(32, 32), 256, 0, stream>>>(Wh[g], oh, ol, 1024);
        }

        // 3) WMMA GEMM + fused LSTM epilogue
        lstm_gemm_packed<<<(BATCH / BM) * (HID / NH), 256, 0, stream>>>(
            Ahi, Alo, Bthi, Btlo, c0,
            b_ii, b_if, b_ig, b_io, b_hi, b_hf, b_hg, b_ho,
            h1, c1);
    } else {
        lstm_cell_wmma_fused<<<(BATCH / BM) * (HID / NH), 256, 0, stream>>>(
            x, h0, c0,
            Wii, b_ii, Wif, b_if, Wig, b_ig, Wio, b_io,
            Whi, b_hi, Whf, b_hf, Whg, b_hg, Who, b_ho,
            h1, c1);
    }
}
